// GridMPNN_45707041964783
// MI455X (gfx1250) — matrix operand order repaired and checked by reading, not compiled, not run on hardware
//
#include <hip/hip_runtime.h>
#include <hip/hip_bf16.h>
#include <math.h>

// ---------------------------------------------------------------------------
// GridMPNN for MI455X (gfx1250), wave32 + WMMA.
//
// m1 factorization: per-edge m1 (258->128) = P[dst] + Q[src] (per-node
// precomputes, stored f16), so only the per-edge m2 GEMM ([E x 128] @
// [128 x 128]) runs per edge, fused with tanh + f32 atomic segment-sum.
//
// Weights are pre-packed once per launch into f16 "B-fragment order":
//   Wf[kt][hi][col][j],  k = kt*32 + (j<8 ? hi*8+j : 16+hi*8+j-8)  (hi = lane>>4, j = 0..15)
// so each lane's 16-half B fragment is one contiguous 32-byte load.
// All node-level GEMMs have M % 16 == 0, so row guards are compiled out.
// Edge grid is split into full (unguarded) blocks + one guarded tail block.
// ---------------------------------------------------------------------------

typedef __attribute__((ext_vector_type(16))) _Float16 v16h;
typedef __attribute__((ext_vector_type(8)))  _Float16 v8h;
typedef __attribute__((ext_vector_type(4)))  _Float16 v4h;
typedef __attribute__((ext_vector_type(8)))  float    v8f;

#define NIN 50000
#define NEX 10000
#define NEDGE 500000
#define HDIM 128
#define EPB 64          // edges per block in the edge-message kernel
#define MAXKPAD 256     // largest padded K in the cat-GEMM (u1: 256)

// Native v_tanh_f32 (TRANS32 op on gfx1250) when the builtin exists.
#if defined(__has_builtin)
#if __has_builtin(__builtin_amdgcn_tanhf)
#define HAVE_NATIVE_TANH 1
#endif
#endif
__device__ __forceinline__ float fast_tanh(float x)
{
#ifdef HAVE_NATIVE_TANH
    return __builtin_amdgcn_tanhf(x);
#else
    float xc = fminf(fmaxf(x, -15.0f), 15.0f);
    float e  = __expf(2.0f * xc);
    return (e - 1.0f) / (e + 1.0f);
#endif
}

// ---------------------------------------------------------------------------
// Weight packer: f32 [K x N] (rows split W0 = [0,d0), W1 = [d0,K)) -> f16
// fragment-ordered [Kpad x 128] with zero padding. All guards live here.
// ---------------------------------------------------------------------------
__global__ void pack_w_kernel(const float* __restrict__ W0,
                              const float* __restrict__ W1,
                              int d0, int K, int ldw, int N,
                              _Float16* __restrict__ Wf, int total)
{
    int idx = blockIdx.x * blockDim.x + threadIdx.x;
    if (idx >= total) return;
    int j  = idx & 15;
    int c  = (idx >> 4) & 127;
    int hi = (idx >> 11) & 1;
    int kt = idx >> 12;
    int k  = kt * 32 + ((j < 8) ? (hi * 8 + j) : (16 + hi * 8 + (j - 8)));
    float v = 0.0f;
    if (c < N) {
        if (k < d0)      v = W0[(size_t)k * ldw + c];
        else if (k < K)  v = W1[(size_t)(k - d0) * ldw + c];
    }
    Wf[idx] = (_Float16)v;
}

// ---------------------------------------------------------------------------
// Cat-input GEMM:  C[M x 128] = act( [A0 | scale1*A1] @ W + bias )
// Block = 256 threads = 8 waves; block owns 16 rows, wave w owns cols
// [16w,16w+16). Requires M % 16 == 0. OUT_F16 stores f16 (for P/Q).
// ---------------------------------------------------------------------------
template <bool HAS_BIAS, bool DO_TANH, bool OUT_F16>
__global__ __launch_bounds__(256) void mlp_cat_wmma_kernel(
    const float* __restrict__ A0, int d0,
    const float* __restrict__ A1, int d1, float scale1,
    const _Float16* __restrict__ Wf,
    const float* __restrict__ bias,
    void* __restrict__ Cv)
{
    __shared__ _Float16 ldsA[16 * (MAXKPAD + 8)];

    const int K    = d0 + d1;
    const int Kpad = (K + 31) & ~31;
    const int lstr = Kpad + 8;
    const int rowBase = blockIdx.x * 16;
    const int tid  = threadIdx.x;

#pragma unroll 1
    for (int r = 0; r < 16; ++r) {
        const int row = rowBase + r;
        for (int k = tid; k < Kpad; k += 256) {
            float v = 0.0f;
            if (k < d0)      v = A0[(size_t)row * d0 + k];
            else if (k < K)  v = scale1 * A1[(size_t)row * d1 + (k - d0)];
            ldsA[r * lstr + k] = (_Float16)v;
        }
    }
    __syncthreads();

    const int lane    = tid & 31;
    const int rl      = lane & 15;
    const int hi      = lane >> 4;
    const int halfSel = hi * 8;
    const int col     = (tid >> 5) * 16 + rl;

    v8f acc = {};
    for (int k0 = 0; k0 < Kpad; k0 += 32) {
        v16h a;
        const _Float16* ap = &ldsA[rl * lstr + k0 + halfSel];
#pragma unroll
        for (int i = 0; i < 8; ++i) { a[i] = ap[i]; a[8 + i] = ap[16 + i]; }
        const v16h b = *(const v16h*)(Wf +
            ((((size_t)(k0 >> 5) * 2 + hi) << 11) + (col << 4)));
        acc = __builtin_amdgcn_wmma_f32_16x16x32_f16(
            false, a, false, b, (short)0, acc, false, false);
    }

    const int rOff = hi * 8;
    const float bc = HAS_BIAS ? bias[col] : 0.0f;
#pragma unroll
    for (int v = 0; v < 8; ++v) {
        const int row = rowBase + v + rOff;
        float y = acc[v] + bc;
        if (DO_TANH) y = fast_tanh(y);
        if (OUT_F16)
            ((_Float16*)Cv)[(size_t)row * HDIM + col] = (_Float16)y;
        else
            ((float*)Cv)[(size_t)row * HDIM + col] = y;
    }
}

// ---------------------------------------------------------------------------
// Square K=128 GEMM:  C[M x N] = act( A @ W + bias ) (+ residual)
// Fully unrolled 4-step K loop, float4 A staging. Requires M % 16 == 0.
// Used for emb2/ex2/u2/out1/out2 (out2: NSMALL -> N = 3, ldc = 3).
// ---------------------------------------------------------------------------
template <bool DO_TANH, bool HAS_RES, bool NSMALL>
__global__ __launch_bounds__(256) void mlp128_wmma_kernel(
    const float* __restrict__ A0,
    const _Float16* __restrict__ Wf,
    const float* __restrict__ bias,
    const float* __restrict__ residual,
    float* __restrict__ C, int ldc)
{
    __shared__ _Float16 ldsA[16 * 136];

    const int rowBase = blockIdx.x * 16;
    const int tid = threadIdx.x;

#pragma unroll
    for (int i = tid; i < 512; i += 256) {
        const int r  = i >> 5;
        const int c4 = (i & 31) << 2;
        const float4 p = *(const float4*)&A0[(size_t)(rowBase + r) * HDIM + c4];
        v4h t;
        t[0] = (_Float16)p.x; t[1] = (_Float16)p.y;
        t[2] = (_Float16)p.z; t[3] = (_Float16)p.w;
        *(v4h*)&ldsA[r * 136 + c4] = t;
    }
    __syncthreads();

    const int lane    = tid & 31;
    const int rl      = lane & 15;
    const int hi      = lane >> 4;
    const int halfSel = hi * 8;
    const int col     = (tid >> 5) * 16 + rl;

    v8f acc = {};
#pragma unroll
    for (int ks = 0; ks < 4; ++ks) {
        v16h a;
        const _Float16* ap = &ldsA[rl * 136 + ks * 32 + halfSel];
#pragma unroll
        for (int i = 0; i < 8; ++i) { a[i] = ap[i]; a[8 + i] = ap[16 + i]; }
        const v16h b = *(const v16h*)(Wf +
            ((((size_t)ks * 2 + hi) << 11) + (col << 4)));
        acc = __builtin_amdgcn_wmma_f32_16x16x32_f16(
            false, a, false, b, (short)0, acc, false, false);
    }

    const int rOff = hi * 8;
    const bool colOk = !NSMALL || (col < 3);
    const float bc = colOk ? bias[col] : 0.0f;
#pragma unroll
    for (int v = 0; v < 8; ++v) {
        const int row = rowBase + v + rOff;
        if (colOk) {
            float y = acc[v] + bc;
            if (DO_TANH) y = fast_tanh(y);
            if (HAS_RES) y += residual[(size_t)row * ldc + col];
            C[(size_t)row * ldc + col] = y;
        }
    }
}

// ---------------------------------------------------------------------------
// Edge message kernel: for each edge e,
//   t = tanh(P[dst[e]] + Q[src[e]])           (b1 folded into P)
//   y = tanh(t @ Wm2 + b2);  agg[dst[e]] += y  (f32 global atomics)
// 64 edges per block; P/Q are f16 (halved gather traffic, L2-resident).
// FULL=true: block is entirely in-range -> no guards anywhere.
// ---------------------------------------------------------------------------
template <bool FULL>
__global__ __launch_bounds__(256) void edge_msg_wmma_kernel(
    const int*      __restrict__ eidx,   // [2E]: [0,E)=src(ext), [E,2E)=dst
    const _Float16* __restrict__ P,      // [NIN x 128] f16 (includes b1)
    const _Float16* __restrict__ Q,      // [NEX x 128] f16
    const _Float16* __restrict__ Wm2f,   // packed [128 x 128]
    const float*    __restrict__ b2,     // [128]
    float*          __restrict__ agg,    // [NIN x 128]
    int eBase, int E)
{
    __shared__ _Float16 Tlds[EPB * 136];
    __shared__ int      srcL[EPB];
    __shared__ int      dstL[EPB];

    const int tid = threadIdx.x;
    const int e0  = eBase + blockIdx.x * EPB;

    if (tid < EPB) {
        const int eg = e0 + tid;
        if (FULL) {
            srcL[tid] = eidx[eg];
            dstL[tid] = eidx[E + eg];
        } else {
            srcL[tid] = (eg < E) ? eidx[eg]     : 0;
            dstL[tid] = (eg < E) ? eidx[E + eg] : 0;
        }
    }
    __syncthreads();

    // Gather P[dst] + Q[src] (16B v8h loads), tanh, store 8 halfs.
#pragma unroll
    for (int i = tid; i < EPB * 16; i += 256) {
        const int e = i >> 4;
        const int c = (i & 15) << 3;
        v8h t = {};
        if (FULL || (e0 + e < E)) {
            const v8h p = *(const v8h*)&P[(size_t)dstL[e] * HDIM + c];
            const v8h q = *(const v8h*)&Q[(size_t)srcL[e] * HDIM + c];
#pragma unroll
            for (int j = 0; j < 8; ++j)
                t[j] = (_Float16)fast_tanh((float)p[j] + (float)q[j]);
        }
        *(v8h*)&Tlds[e * 136 + c] = t;
    }
    __syncthreads();

    const int lane    = tid & 31;
    const int rl      = lane & 15;
    const int hi      = lane >> 4;
    const int halfSel = hi * 8;
    const int col     = (tid >> 5) * 16 + rl;
    const int rOff    = hi * 8;
    const float bc    = b2[col];

#pragma unroll 1
    for (int rt = 0; rt < EPB / 16; ++rt) {
        v8f acc = {};
        const int rowA = rt * 16 + rl;
#pragma unroll
        for (int ks = 0; ks < 4; ++ks) {
            v16h a;
            const _Float16* ap = &Tlds[rowA * 136 + ks * 32 + halfSel];
#pragma unroll
            for (int i = 0; i < 8; ++i) { a[i] = ap[i]; a[8 + i] = ap[16 + i]; }
            const v16h b = *(const v16h*)(Wm2f +
                ((((size_t)ks * 2 + hi) << 11) + (col << 4)));
            acc = __builtin_amdgcn_wmma_f32_16x16x32_f16(
                false, a, false, b, (short)0, acc, false, false);
        }
        // Preload the 8 consecutive dst indices (vectorizable LDS reads),
        // then scatter with straight-line atomics on the FULL path.
        const int eL0 = rt * 16 + rOff;
        int dsts[8];
#pragma unroll
        for (int v = 0; v < 8; ++v) dsts[v] = dstL[eL0 + v];
#pragma unroll
        for (int v = 0; v < 8; ++v) {
            if (FULL || (e0 + eL0 + v < E)) {
                float y = fast_tanh(acc[v] + bc);
                atomicAdd(&agg[(size_t)dsts[v] * HDIM + col], y);
            }
        }
    }
}

// --------------------------- small support kernels -------------------------

__global__ void edge_count_kernel(const int* __restrict__ eidx,
                                  float* __restrict__ cnt, int E)
{
    int e = blockIdx.x * blockDim.x + threadIdx.x;
    if (e < E) atomicAdd(&cnt[eidx[E + e]], 1.0f);
}

__global__ void mean_div_kernel(float* __restrict__ agg,
                                const float* __restrict__ cnt, int n)
{
    int i = blockIdx.x * blockDim.x + threadIdx.x;
    if (i < n) agg[i] = agg[i] / fmaxf(cnt[i >> 7], 1.0f);
}

// Per-channel sum / sumsq: block owns 512 rows, thread column = tid&127.
__global__ __launch_bounds__(256) void colstat_kernel(
    const float* __restrict__ x, float* __restrict__ sum,
    float* __restrict__ sumsq, int M)
{
    const int tid  = threadIdx.x;
    const int c    = tid & 127;
    const int half = tid >> 7;
    int r    = blockIdx.x * 512 + half;
    int rend = blockIdx.x * 512 + 512;
    if (rend > M) rend = M;
    float s = 0.0f, s2 = 0.0f;
    for (; r < rend; r += 2) {
        float v = x[(size_t)r * HDIM + c];
        s += v; s2 += v * v;
    }
    atomicAdd(&sum[c], s);
    atomicAdd(&sumsq[c], s2);
}

__global__ void stat_finalize_kernel(const float* __restrict__ sum,
                                     const float* __restrict__ sumsq,
                                     float* __restrict__ mu,
                                     float* __restrict__ rs, int M)
{
    int c = threadIdx.x;
    if (c < HDIM) {
        float m   = sum[c] / (float)M;
        float var = sumsq[c] / (float)M - m * m;
        mu[c] = m;
        rs[c] = rsqrtf(var + 1e-5f);
    }
}

__global__ void normalize_kernel(float* __restrict__ x,
                                 const float* __restrict__ mu,
                                 const float* __restrict__ rs, int n)
{
    int i = blockIdx.x * blockDim.x + threadIdx.x;
    if (i < n) { int c = i & 127; x[i] = (x[i] - mu[c]) * rs[c]; }
}

// ------------------------------- host side ---------------------------------

extern "C" void kernel_launch(void* const* d_in, const int* in_sizes, int n_in,
                              void* d_out, int out_size, void* d_ws, size_t ws_size,
                              hipStream_t stream)
{
    (void)in_sizes; (void)n_in; (void)out_size; (void)ws_size;

    // setup_inputs() dict order:
    const float* nodes    = (const float*)d_in[0];   // [NIN x 16]
    const float* node_pos = (const float*)d_in[1];   // [NIN x 2]
    /* d_in[2] = edge_index (unused by the math) */
    const float* ex_nodes = (const float*)d_in[3];   // [NEX x 8]
    const float* ex_pos   = (const float*)d_in[4];   // [NEX x 2]
    const int*   eix      = (const int*)d_in[5];     // [2 x E]
    /* d_in[6] = batch (single graph, unused) */
    // params leaves in dict order starting at index 7:
    //  7 emb1.W  8 emb1.b  9 emb2.W 10 emb2.b
    // 11.. g1: ex1.W,b ex2.W,b m1.W,b m2.W,b u1.W,b u2.W,b   (12 leaves)
    // 23.. g2: same; 35 out1.W 36 out1.b 37 out2.W 38 out2.b
    #define PF(i) ((const float*)d_in[(i)])

    float* ws   = (float*)d_ws;
    float* hA   = ws;                        // [NIN x 128] node state
    float* hB   = hA  + (size_t)NIN * HDIM;  // [NIN x 128] scratch
    float* AGG  = hB  + (size_t)NIN * HDIM;  // [NIN x 128] segment sum / mean
    float* EXH  = AGG + (size_t)NIN * HDIM;  // [NEX x 128]
    float* EXH2 = EXH + (size_t)NEX * HDIM;  // [NEX x 128]
    float* cnt  = EXH2+ (size_t)NEX * HDIM;  // [NIN]
    float* ssum = cnt + NIN;                 // [128]
    float* ssq  = ssum + HDIM;               // [128]
    float* muv  = ssq  + HDIM;               // [128]
    float* rsv  = muv  + HDIM;               // [128]

    // f16 arena: P/Q activations + packed weights.
    _Float16* harena = (_Float16*)(rsv + HDIM);
    _Float16* Pf = harena;                         // [NIN x 128] f16
    _Float16* Qf = Pf + (size_t)NIN * HDIM;        // [NEX x 128] f16
    _Float16* wf = Qf + (size_t)NEX * HDIM;
    auto take = [&](int Kpad) { _Float16* p = wf; wf += (size_t)Kpad * 128; return p; };
    auto pack = [&](const float* W0, const float* W1, int d0, int K, int ldw,
                    int N, _Float16* dst) {
        int total = ((K + 31) & ~31) * 128;
        pack_w_kernel<<<dim3((total + 255) / 256), dim3(256), 0, stream>>>(
            W0, W1, d0, K, ldw, N, dst, total);
    };

    _Float16* wEmb1 = take(32);   // K=18
    _Float16* wEmb2 = take(128);
    _Float16* wEx1[2]; _Float16* wEx2[2]; _Float16* wP[2]; _Float16* wQ[2];
    _Float16* wM2[2]; _Float16* wU1[2]; _Float16* wU2[2];
    for (int l = 0; l < 2; ++l) {
        wEx1[l] = take(32);   // K=10
        wEx2[l] = take(128);
        wP[l]   = take(160);  // K=130
        wQ[l]   = take(160);  // K=130
        wM2[l]  = take(128);
        wU1[l]  = take(256);
        wU2[l]  = take(128);
    }
    _Float16* wOut1 = take(128);
    _Float16* wOut2 = take(128);  // N=3, zero-padded columns

    // ---- pack every weight once (cheap, deterministic) ----
    pack(PF(7), PF(7) + 16 * HDIM, 16, 18, HDIM, HDIM, wEmb1);
    pack(PF(9), PF(9), 128, 128, HDIM, HDIM, wEmb2);
    for (int l = 0; l < 2; ++l) {
        const int pb = 11 + l * 12;
        pack(PF(pb + 0), PF(pb + 0) + 8 * HDIM, 8, 10, HDIM, HDIM, wEx1[l]);
        pack(PF(pb + 2), PF(pb + 2), 128, 128, HDIM, HDIM, wEx2[l]);
        // P weights: m1.W rows {0..127} then {256..257}
        pack(PF(pb + 4), PF(pb + 4) + 256 * HDIM, 128, 130, HDIM, HDIM, wP[l]);
        // Q weights: m1.W rows {128..255} then {256..257}
        pack(PF(pb + 4) + 128 * HDIM, PF(pb + 4) + 256 * HDIM, 128, 130, HDIM,
             HDIM, wQ[l]);
        pack(PF(pb + 6), PF(pb + 6), 128, 128, HDIM, HDIM, wM2[l]);
        pack(PF(pb + 8), PF(pb + 8) + 128 * HDIM, 128, 256, HDIM, HDIM, wU1[l]);
        pack(PF(pb +10), PF(pb +10), 128, 128, HDIM, HDIM, wU2[l]);
    }
    pack(PF(35), PF(35), 128, 128, HDIM, HDIM, wOut1);
    pack(PF(37), PF(37), 128, 128, 3, 3, wOut2);

    const dim3 B256(256);
    const dim3 gNin(NIN / 16), gNex(NEX / 16);
    const int fullBlocks = NEDGE / EPB;                 // 7812
    const int tailEdges  = NEDGE - fullBlocks * EPB;    // 32

    // ---- embedding: h = tanh(emb2(tanh(emb1(cat(nodes, pos))))) ----
    mlp_cat_wmma_kernel<true, true, false><<<gNin, B256, 0, stream>>>(
        nodes, 16, node_pos, 2, 1.0f, wEmb1, PF(8), hB);
    mlp128_wmma_kernel<true, false, false><<<gNin, B256, 0, stream>>>(
        hB, wEmb2, PF(10), nullptr, hA, HDIM);

    // ---- two GNN layers ----
    for (int layer = 0; layer < 2; ++layer) {
        const int pb = 11 + layer * 12;

        // External embed
        mlp_cat_wmma_kernel<true, true, false><<<gNex, B256, 0, stream>>>(
            ex_nodes, 8, ex_pos, 2, 1.0f, wEx1[layer], PF(pb + 1), EXH);
        mlp128_wmma_kernel<true, false, false><<<gNex, B256, 0, stream>>>(
            EXH, wEx2[layer], PF(pb + 3), nullptr, EXH2, HDIM);

        // m1 factorization (f16 outputs):
        //   P = h@Wa + pos@Wc + b1 ;  Q = exh@Wb - ex_pos@Wc
        mlp_cat_wmma_kernel<true, false, true><<<gNin, B256, 0, stream>>>(
            hA, HDIM, node_pos, 2, 1.0f, wP[layer], PF(pb + 5), Pf);
        mlp_cat_wmma_kernel<false, false, true><<<gNex, B256, 0, stream>>>(
            EXH2, HDIM, ex_pos, 2, -1.0f, wQ[layer], nullptr, Qf);

        // Segment mean accumulators
        hipMemsetAsync(AGG, 0, (size_t)NIN * HDIM * sizeof(float), stream);
        hipMemsetAsync(cnt, 0, (size_t)NIN * sizeof(float), stream);
        edge_count_kernel<<<dim3((NEDGE + 255) / 256), B256, 0, stream>>>(
            eix, cnt, NEDGE);

        // Per-edge: tanh(P+Q) -> WMMA @ Wm2 -> tanh -> atomic scatter
        edge_msg_wmma_kernel<true><<<dim3(fullBlocks), B256, 0, stream>>>(
            eix, Pf, Qf, wM2[layer], PF(pb + 7), AGG, 0, NEDGE);
        if (tailEdges)
            edge_msg_wmma_kernel<false><<<dim3(1), B256, 0, stream>>>(
                eix, Pf, Qf, wM2[layer], PF(pb + 7), AGG, fullBlocks * EPB,
                NEDGE);

        // mean = agg / max(cnt, 1)
        mean_div_kernel<<<dim3((NIN * HDIM + 255) / 256), B256, 0, stream>>>(
            AGG, cnt, NIN * HDIM);

        // update: u = tanh(u2(tanh(u1(cat(h, mean)))));  x = h + u (in place)
        mlp_cat_wmma_kernel<true, true, false><<<gNin, B256, 0, stream>>>(
            hA, HDIM, AGG, HDIM, 1.0f, wU1[layer], PF(pb + 9), hB);
        mlp128_wmma_kernel<true, true, false><<<gNin, B256, 0, stream>>>(
            hB, wU2[layer], PF(pb + 11), hA, hA, HDIM);

        // InstanceNorm (affine=False, biased var) in place on hA
        hipMemsetAsync(ssum, 0, 2 * HDIM * sizeof(float), stream);
        colstat_kernel<<<dim3((NIN + 511) / 512), B256, 0, stream>>>(
            hA, ssum, ssq, NIN);
        stat_finalize_kernel<<<dim3(1), dim3(HDIM), 0, stream>>>(
            ssum, ssq, muv, rsv, NIN);
        normalize_kernel<<<dim3((NIN * HDIM + 255) / 256), B256, 0, stream>>>(
            hA, muv, rsv, NIN * HDIM);
    }

    // ---- readout: out = out2(tanh(out1(h))) ----
    mlp128_wmma_kernel<true, false, false><<<gNin, B256, 0, stream>>>(
        hA, wOut1, PF(36), nullptr, hB, HDIM);
    mlp128_wmma_kernel<false, false, true><<<gNin, B256, 0, stream>>>(
        hB, wOut2, PF(38), nullptr, (float*)d_out, 3);
    #undef PF
}
